// CrossAttention_G_49014166782304
// MI455X (gfx1250) — matrix operand order repaired and checked by reading, not compiled, not run on hardware
//
#include <hip/hip_runtime.h>
#include <hip/hip_bf16.h>
#include <math.h>

typedef __attribute__((ext_vector_type(16))) _Float16 v16h;
typedef __attribute__((ext_vector_type(8)))  float    v8f;
typedef __attribute__((ext_vector_type(4)))  unsigned int v4u;
typedef __attribute__((ext_vector_type(8)))  int      v8i;
typedef __attribute__((ext_vector_type(4)))  int      v4i;

#define B_      8
#define N_      1872
#define D_      768
#define NTILES  117          /* N_/16 */
#define KCH     24           /* D_/32 */
#define NPAD    1888         /* 59*32 : K-padded score width */
#define KCH2    59           /* NPAD/32 */
#define AT_THREADS 128

#define SS_BYTES   ((size_t)16 * NPAD * 4)   /* 120832 */
#define SX_BYTES   ((size_t)16 * D_ * 2)     /* 24576  */
#define SK_BYTES   ((size_t)16 * D_ * 2)     /* 24576 per wave */
#define LDS_BYTES  (SS_BYTES + SX_BYTES + 4 * SK_BYTES)

// ---------------------------------------------------------------------------
// Async global->LDS copy of 16 bytes per lane (ASYNCcnt-tracked)
// ---------------------------------------------------------------------------
__device__ __forceinline__ void async_g2l_b128(unsigned lds_off, const void* g) {
    asm volatile("global_load_async_to_lds_b128 %0, %1, off"
                 :: "v"(lds_off), "v"(g) : "memory");
}
__device__ __forceinline__ void wait_asynccnt0() {
    asm volatile("s_wait_asynccnt 0x0" ::: "memory");
}

// ---------------------------------------------------------------------------
// Tensor Data Mover: 2D tile (f16 elements) global -> LDS
// (6-arg builtin form: g0, g1, g2, g3, g4, cpol)
// ---------------------------------------------------------------------------
__device__ __forceinline__ void tdm_load_tile_f16(
        unsigned lds_off, const void* gptr,
        unsigned tensor_w, unsigned tensor_h,
        unsigned tile_w, unsigned tile_h, unsigned stride_elems) {
    unsigned long long ga = (unsigned long long)(uintptr_t)gptr;
    v4u g0;
    g0[0] = 1u;                                        // count=1 (valid), user mode
    g0[1] = lds_off;                                   // lds_addr [63:32]
    g0[2] = (unsigned)(ga & 0xFFFFFFFFu);              // global_addr [95:64]
    g0[3] = (unsigned)((ga >> 32) & 0x01FFFFFFu)       // global_addr [120:96]
          | (2u << 30);                                // type=2 ("image") [127:126]
    v8i g1;
    g1[0] = (int)(1u << 16);                           // data_size=1 -> 2 bytes
    g1[1] = (int)((tensor_w & 0xFFFFu) << 16);         // tensor_dim0[15:0] @ 63:48
    g1[2] = (int)(((tensor_w >> 16) & 0xFFFFu)         // tensor_dim0[31:16]
          | ((tensor_h & 0xFFFFu) << 16));             // tensor_dim1[15:0]
    g1[3] = (int)(((tensor_h >> 16) & 0xFFFFu)         // tensor_dim1[31:16]
          | ((tile_w & 0xFFFFu) << 16));               // tile_dim0 @ 127:112
    g1[4] = (int)(tile_h & 0xFFFFu);                   // tile_dim1; tile_dim2=0
    g1[5] = (int)stride_elems;                         // tensor_dim0_stride[31:0]
    g1[6] = 0;                                         // stride hi / dim1_stride lo
    g1[7] = 0;                                         // dim1_stride hi
    v4i z4 = {0, 0, 0, 0};
    v8i z8 = {0, 0, 0, 0, 0, 0, 0, 0};
    __builtin_amdgcn_tensor_load_to_lds(g0, g1, z4, z4, z8, 0);
}

// ---------------------------------------------------------------------------
// f32 -> f16 conversion (grid-stride)
// ---------------------------------------------------------------------------
__global__ void cvt_f32_f16(const float* __restrict__ src, _Float16* __restrict__ dst, int n) {
    int i = blockIdx.x * blockDim.x + threadIdx.x;
    int stride = gridDim.x * blockDim.x;
    for (; i < n; i += stride) dst[i] = (_Float16)src[i];
}

// ---------------------------------------------------------------------------
// k = y @ W^T + b    (one wave per block; 16 rows x 64 cols per block)
// grid: (B_*NTILES, D_/64)
// ---------------------------------------------------------------------------
__global__ void __launch_bounds__(32) gemm_k_kernel(
        const _Float16* __restrict__ yb, const _Float16* __restrict__ Wb,
        const float* __restrict__ bias, _Float16* __restrict__ kb) {
    int bx = blockIdx.x;
    int b  = bx / NTILES, nt = bx % NTILES;
    int ocbase = blockIdx.y * 64;
    int lane = threadIdx.x;
    int mrow = lane & 15, half = lane >> 4;

    const _Float16* Yrow = yb + ((size_t)b * N_ + (size_t)nt * 16) * D_;
    v8f acc[4] = {};

    for (int kc = 0; kc < KCH; ++kc) {
        v16h a;                                   // A 16x32 (ISA 7.12.2 layout)
#pragma unroll
        for (int v = 0; v < 8; ++v) {
            int kik = ((v >= 4) ? 16 : 0) + half * 8 + (v & 3) * 2;
            const _Float16* p = Yrow + (size_t)mrow * D_ + kc * 32 + kik;
            a[2 * v] = p[0];  a[2 * v + 1] = p[1];
        }
#pragma unroll
        for (int j = 0; j < 4; ++j) {
            int o = ocbase + j * 16 + mrow;
            v16h bb;
#pragma unroll
            for (int v = 0; v < 8; ++v) {
                int i = kc * 32 + ((v >= 4) ? 16 : 0) + half * 8 + (v & 3) * 2;
                const _Float16* p = Wb + (size_t)o * D_ + i;
                bb[2 * v] = p[0];  bb[2 * v + 1] = p[1];
            }
            acc[j] = __builtin_amdgcn_wmma_f32_16x16x32_f16(
                         false, a, false, bb, (short)0, acc[j], false, false);
        }
    }
#pragma unroll
    for (int j = 0; j < 4; ++j) {
        int o = ocbase + j * 16 + mrow;
        float bv = bias[o];
#pragma unroll
        for (int r = 0; r < 8; ++r) {
            int row = nt * 16 + r + half * 8;
            kb[((size_t)b * N_ + row) * D_ + o] = (_Float16)(acc[j][r] + bv);
        }
    }
}

// ---------------------------------------------------------------------------
// Fused attention: per (batch, 16-row tile)
// ---------------------------------------------------------------------------
__global__ void __launch_bounds__(AT_THREADS) attn_kernel(
        const _Float16* __restrict__ xh, const _Float16* __restrict__ yb,
        const _Float16* __restrict__ kb, const float* __restrict__ rel,
        const float* __restrict__ focus_p, const float* __restrict__ gating_p,
        const float* __restrict__ temp_p,
        float* __restrict__ out, float* __restrict__ heat) {
    extern __shared__ char smem[];
    float*    sS = (float*)smem;                         // [16][NPAD]
    _Float16* sX = (_Float16*)(smem + SS_BYTES);         // [16][D_] x tile (f16)

    int blk = blockIdx.x;
    int b = blk / NTILES, nt = blk % NTILES;
    int tid = threadIdx.x;
    int wave = tid >> 5, lane = tid & 31;
    int mrow = lane & 15, half = lane >> 4;

    _Float16* sKw = (_Float16*)(smem + SS_BYTES + SX_BYTES + (size_t)wave * SK_BYTES);

    // ---- stage x row tile into LDS via Tensor Data Mover ---------------
    if (wave == 0) {
        const _Float16* xt = xh + ((size_t)b * N_ + (size_t)nt * 16) * D_;
        tdm_load_tile_f16((unsigned)(uintptr_t)sX, xt,
                          D_, N_, D_, 16, D_);
        __builtin_amdgcn_s_wait_tensorcnt(0);
    }
    __syncthreads();

    const float scale = 0.03608439182435161f;  // 1/sqrt(768)
    const _Float16* Kb = kb + (size_t)b * N_ * D_;
    unsigned sKw_off = (unsigned)(uintptr_t)sKw;

    // ---- phase 1: score slab S = x k^T / sqrt(D) -----------------------
    for (int ct = wave; ct < NTILES; ct += 4) {
        // async-stage this wave's 16 contiguous k rows (24576 B) into LDS
        const char* kt = (const char*)(Kb + (size_t)ct * 16 * D_);
        __builtin_prefetch(kt + (size_t)4 * 16 * D_ * 2, 0, 3);   // next tile
        for (int i = lane; i < (16 * D_ * 2) / 16; i += 32)
            async_g2l_b128(sKw_off + (unsigned)i * 16u, kt + (size_t)i * 16);
        wait_asynccnt0();

        v8f c = {};
        for (int kc = 0; kc < KCH; ++kc) {
            v16h a;
#pragma unroll
            for (int v = 0; v < 8; ++v) {
                int kik = ((v >= 4) ? 16 : 0) + half * 8 + (v & 3) * 2;
                const _Float16* p = sX + (size_t)mrow * D_ + kc * 32 + kik;
                a[2 * v] = p[0];  a[2 * v + 1] = p[1];
            }
            v16h bb;                                  // B[d][m] = k[m][d] from LDS
#pragma unroll
            for (int v = 0; v < 8; ++v) {
                const _Float16* p = sKw + (size_t)mrow * D_ + kc * 32 + ((v >= 4) ? 16 : 0) + half * 8 + (v & 3) * 2;
                bb[2 * v] = p[0];  bb[2 * v + 1] = p[1];
            }
            c = __builtin_amdgcn_wmma_f32_16x16x32_f16(
                    false, a, false, bb, (short)0, c, false, false);
        }
#pragma unroll
        for (int r = 0; r < 8; ++r)
            sS[(size_t)(r + half * 8) * NPAD + ct * 16 + mrow] = c[r] * scale;
    }
    __syncthreads();

    // ---- phase 2: dual softmax + gate + entropy ------------------------
    {
        float gate = 1.f / (1.f + expf(-gating_p[0]));
        float foc  = fabsf(focus_p[0]);
        int r = tid >> 3;            // row 0..15, 8 threads per row
        int tsub = tid & 7;
        int rowg = nt * 16 + r;
        const float* relrow = rel + (size_t)rowg * N_;
        float* srow = sS + (size_t)r * NPAD;

        float mx = -1e30f, Z = 0.f;
        for (int m = tsub; m < N_; m += 8) {
            mx = fmaxf(mx, srow[m]);
            float dy = relrow[m];
            Z += expf(-foc * dy * dy);
        }
        for (int off = 4; off; off >>= 1) {
            mx = fmaxf(mx, __shfl_xor(mx, off, 8));
            Z += __shfl_xor(Z, off, 8);
        }
        float se = 0.f;
        for (int m = tsub; m < N_; m += 8) se += expf(srow[m] - mx);
        for (int off = 4; off; off >>= 1) se += __shfl_xor(se, off, 8);

        float wp = (1.f - gate) / se;
        float wq = gate / Z;         // rows of attn already sum to 1
        float ent = 0.f;
        for (int m = tsub; m < N_; m += 8) {
            float dy = relrow[m];
            float a = expf(srow[m] - mx) * wp + expf(-foc * dy * dy) * wq;
            srow[m] = a;
            ent += -a * logf(a + 1e-8f);
        }
        for (int off = 4; off; off >>= 1) ent += __shfl_xor(ent, off, 8);
        for (int m = N_ + tsub; m < NPAD; m += 8) srow[m] = 0.f;  // zero K-pad
        if (tsub == 0) {
            float t = temp_p[0];
            heat[(size_t)b * N_ + rowg] = 2.f - 2.f / (1.f + expf(-t * ent));
        }
    }
    __syncthreads();

    // ---- phase 3: out = attn @ y  (each wave owns 192 output cols) -----
    const _Float16* Yb = yb + (size_t)b * N_ * D_;
    int obase = wave * 192;
    v8f acc[12] = {};
    for (int kc = 0; kc < KCH2; ++kc) {
        v16h a;
#pragma unroll
        for (int v = 0; v < 8; ++v) {
            int kik = ((v >= 4) ? 16 : 0) + half * 8 + (v & 3) * 2;
            int m = kc * 32 + kik;
            a[2 * v]     = (_Float16)sS[(size_t)mrow * NPAD + m];
            a[2 * v + 1] = (_Float16)sS[(size_t)mrow * NPAD + m + 1];
        }
#pragma unroll
        for (int j = 0; j < 12; ++j) {
            int o = obase + j * 16 + mrow;
            v16h bb;                                 // B[m][o] = y[m][o]
#pragma unroll
            for (int v = 0; v < 8; ++v) {
                int m0 = kc * 32 + ((v >= 4) ? 16 : 0) + half * 8 + (v & 3) * 2;
                int m1 = m0 + 1;
                if (m0 >= N_) m0 = N_ - 1;           // pad rows: attn==0
                if (m1 >= N_) m1 = N_ - 1;
                bb[2 * v]     = Yb[(size_t)m0 * D_ + o];
                bb[2 * v + 1] = Yb[(size_t)m1 * D_ + o];
            }
            acc[j] = __builtin_amdgcn_wmma_f32_16x16x32_f16(
                         false, a, false, bb, (short)0, acc[j], false, false);
        }
    }
#pragma unroll
    for (int j = 0; j < 12; ++j) {
        int o = obase + j * 16 + mrow;
#pragma unroll
        for (int r = 0; r < 8; ++r) {
            int row = nt * 16 + r + half * 8;
            out[((size_t)b * N_ + row) * D_ + o] = acc[j][r];
        }
    }
}

// ---------------------------------------------------------------------------
extern "C" void kernel_launch(void* const* d_in, const int* in_sizes, int n_in,
                              void* d_out, int out_size, void* d_ws, size_t ws_size,
                              hipStream_t stream) {
    (void)in_sizes; (void)n_in; (void)out_size; (void)ws_size;
    const float* x      = (const float*)d_in[0];
    const float* y      = (const float*)d_in[1];
    const float* W      = (const float*)d_in[2];
    const float* bias   = (const float*)d_in[3];
    const float* focus  = (const float*)d_in[4];
    const float* gating = (const float*)d_in[5];
    const float* temp   = (const float*)d_in[6];
    const float* rel    = (const float*)d_in[7];

    float* out  = (float*)d_out;
    float* heat = out + (size_t)B_ * N_ * D_;

    char* ws = (char*)d_ws;
    size_t nYb = (size_t)B_ * N_ * D_ * 2;       // f16 y
    size_t nWb = (size_t)D_ * D_ * 2;            // f16 W
    _Float16* yb = (_Float16*)ws;
    _Float16* Wb = (_Float16*)(ws + nYb);
    _Float16* kb = (_Float16*)(ws + nYb + nWb);
    _Float16* xhp = (_Float16*)(ws + nYb + nWb + nYb);

    int nY = B_ * N_ * D_;
    cvt_f32_f16<<<(nY + 255) / 256, 256, 0, stream>>>(y, yb, nY);
    cvt_f32_f16<<<(nY + 255) / 256, 256, 0, stream>>>(x, xhp, nY);
    int nW = D_ * D_;
    cvt_f32_f16<<<(nW + 255) / 256, 256, 0, stream>>>(W, Wb, nW);

    dim3 gK(B_ * NTILES, D_ / 64);
    gemm_k_kernel<<<gK, 32, 0, stream>>>(yb, Wb, bias, kb);

    attn_kernel<<<B_ * NTILES, AT_THREADS, LDS_BYTES, stream>>>(
        xhp, yb, kb, rel, focus, gating, temp, out, heat);
}
